// BlenderbotMultiHeadAttention_85933705659014
// MI455X (gfx1250) — matrix operand order repaired and checked by reading, not compiled, not run on hardware
//
#include <hip/hip_runtime.h>

#define B_  2
#define S_  2048
#define D_  1024
#define H_  16
#define DH_ 64
#define M_TOT (B_*S_)   // 4096

typedef __attribute__((ext_vector_type(16))) __bf16 v16bf;
typedef __attribute__((ext_vector_type(8)))  float  v8f;

union FragAB {
  v16bf v;
  uint4 q[2];
};

__device__ __forceinline__ unsigned f2bf(float f) {
  union { float f; unsigned u; } x; x.f = f;
  unsigned b = x.u;
  return (b + 0x7FFFu + ((b >> 16) & 1u)) >> 16;
}

__device__ __forceinline__ float rmax16(float v) {
  v = fmaxf(v, __shfl_xor(v, 1, 32));
  v = fmaxf(v, __shfl_xor(v, 2, 32));
  v = fmaxf(v, __shfl_xor(v, 4, 32));
  v = fmaxf(v, __shfl_xor(v, 8, 32));
  return v;
}
__device__ __forceinline__ float rsum16(float v) {
  v += __shfl_xor(v, 1, 32);
  v += __shfl_xor(v, 2, 32);
  v += __shfl_xor(v, 4, 32);
  v += __shfl_xor(v, 8, 32);
  return v;
}

__device__ __forceinline__ v8f wmma_bf16(const FragAB& a, const FragAB& b, v8f c) {
  return __builtin_amdgcn_wmma_f32_16x16x32_bf16(false, a.v, false, b.v,
                                                 (short)0, c, false, false);
}

// ---------------------------------------------------------------- f32 -> bf16
__global__ void cvt_f32_bf16(const float* __restrict__ src,
                             unsigned short* __restrict__ dst, int n) {
  int i = (blockIdx.x * blockDim.x + threadIdx.x) * 8;
  if (i + 7 < n) {
    const float4 f0 = *(const float4*)(src + i);
    const float4 f1 = *(const float4*)(src + i + 4);
    uint4 r;
    r.x = f2bf(f0.x) | (f2bf(f0.y) << 16);
    r.y = f2bf(f0.z) | (f2bf(f0.w) << 16);
    r.z = f2bf(f1.x) | (f2bf(f1.y) << 16);
    r.w = f2bf(f1.z) | (f2bf(f1.w) << 16);
    *(uint4*)(dst + i) = r;
  }
}

// -------------------------------------------------- 32x64 GEMM core
// acc[2][4] covers a 32(M) x 64(N) tile. K loop FULLY unrolled (32 chunks):
// no loop-carried PHIs on acc (no v_mov copies / hazard nops), and the
// scheduler is free to hoist each chunk's b128 loads over earlier WMMAs.
__device__ __forceinline__ void gemm_acc_32x64(const unsigned short* __restrict__ X,
                                               const unsigned short* __restrict__ W,
                                               int tm, int tn, int lane,
                                               v8f (&acc)[2][4]) {
  const int lo16 = lane & 15;
  const int hi   = lane >> 4;
  const unsigned short* a0p = X + (tm * 32 + lo16) * D_ + hi * 8;
  const unsigned short* a1p = a0p + 16 * D_;
  const unsigned short* bb  = W + (tn * 64 + lo16) * D_ + hi * 8;

#pragma unroll
  for (int k = 0; k < D_; k += 32) {
    FragAB a0, a1;
    a0.q[0] = *(const uint4*)(a0p + k);
    a0.q[1] = *(const uint4*)(a0p + k + 16);
    a1.q[0] = *(const uint4*)(a1p + k);
    a1.q[1] = *(const uint4*)(a1p + k + 16);
#pragma unroll
    for (int j = 0; j < 4; ++j) {
      FragAB b;
      b.q[0] = *(const uint4*)(bb + j * 16 * D_ + k);
      b.q[1] = *(const uint4*)(bb + j * 16 * D_ + k + 16);
      acc[0][j] = wmma_bf16(a0, b, acc[0][j]);
      acc[1][j] = wmma_bf16(a1, b, acc[1][j]);
    }
  }
}

// ---------------------------------------------------------------- QKV GEMM
// out = X(4096x1024,bf16) @ W^T + b, gridDim.y: 0=Q 1=K 2=V.
// Q,K -> [B,H,S,DH] bf16 (Q scaled 1/8); V -> [B,H,DH,S] bf16 (transposed).
__global__ void __launch_bounds__(256, 1)
gemm_qkv(const unsigned short* __restrict__ X,
         const unsigned short* __restrict__ Wq,
         const unsigned short* __restrict__ Wk,
         const unsigned short* __restrict__ Wv,
         const float* __restrict__ bq,
         const float* __restrict__ bk,
         const float* __restrict__ bv,
         unsigned short* __restrict__ Qh,
         unsigned short* __restrict__ Kh,
         unsigned short* __restrict__ Vt) {
  const int mat = blockIdx.y;
  const unsigned short* W = (mat == 0) ? Wq : (mat == 1) ? Wk : Wv;
  const float* bias = (mat == 0) ? bq : (mat == 1) ? bk : bv;

  const int wave = (blockIdx.x * blockDim.x + threadIdx.x) >> 5;
  const int lane = threadIdx.x & 31;
  const int lo16 = lane & 15;
  const int hi   = lane >> 4;
  const int tn = wave & 15;          // 16 tiles of 64 cols
  const int tm = wave >> 4;          // 128 tiles of 32 rows

  v8f acc[2][4] = {};
  gemm_acc_32x64(X, W, tm, tn, lane, acc);

  const float scale = (mat == 0) ? 0.125f : 1.0f;
#pragma unroll
  for (int mi = 0; mi < 2; ++mi) {
#pragma unroll
    for (int j = 0; j < 4; ++j) {
      const int n  = tn * 64 + j * 16 + lo16;
      const float bsc = bias[n];
      const int h = n >> 6, dh = n & 63;
#pragma unroll
      for (int r = 0; r < 8; ++r) {
        const int m = tm * 32 + mi * 16 + r + (hi << 3);
        const int b = m >> 11, s = m & (S_ - 1);
        const unsigned short bfv =
            (unsigned short)f2bf((acc[mi][j][r] + bsc) * scale);
        if (mat == 2)
          Vt[(((b * H_ + h) * DH_) + dh) * S_ + s] = bfv;
        else if (mat == 0)
          Qh[(((b * H_ + h) * S_) + s) * DH_ + dh] = bfv;
        else
          Kh[(((b * H_ + h) * S_) + s) * DH_ + dh] = bfv;
      }
    }
  }
}

// ---------------------------------------------------------------- flash attn
// One wave per (bh, 16-row q tile). Keys streamed 32 at a time, K-fragments
// ping-pong double-buffered; V loads issued before softmax so VALU hides them.
__global__ void __launch_bounds__(256, 1)
flash_attn(const unsigned short* __restrict__ Qh,
           const unsigned short* __restrict__ Kh,
           const unsigned short* __restrict__ Vt,
           unsigned short* __restrict__ Ctx) {
  __shared__ unsigned short lds_p[8][16 * 32];

  const int wid  = threadIdx.x >> 5;
  const int lane = threadIdx.x & 31;
  const int lo16 = lane & 15;
  const int hi   = lane >> 4;
  const int wave = blockIdx.x * (blockDim.x >> 5) + wid;
  const int bh = wave >> 7;              // 128 q-tiles per head-batch
  const int q0 = (wave & 127) * 16;
  const int krun = hi * 8;

  // Q A-fragments: 16 rows x 64 dh -> two 16x32 fragments, kept in registers
  const unsigned short* qbase = Qh + (bh * S_ + q0 + lo16) * DH_;
  FragAB aq0, aq1;
  aq0.q[0] = *(const uint4*)(qbase + krun);
  aq0.q[1] = *(const uint4*)(qbase + krun + 16);
  aq1.q[0] = *(const uint4*)(qbase + 32 + krun);
  aq1.q[1] = *(const uint4*)(qbase + 32 + krun + 16);

  float mrow[8], lrow[8];
#pragma unroll
  for (int r = 0; r < 8; ++r) { mrow[r] = -__builtin_inff(); lrow[r] = 0.f; }
  v8f acc[4] = {};

  const unsigned short* kbh = Kh + bh * S_ * DH_;
  const unsigned short* vbh = Vt + bh * DH_ * S_;

  // kf[buf][0..3] = {keys 0-15 x dh 0-31, keys 0-15 x dh 32-63,
  //                  keys 16-31 x dh 0-31, keys 16-31 x dh 32-63}
  FragAB kf[2][4];
  auto load_k = [&](int buf, int kb) {
    const unsigned short* kp0 = kbh + (kb + lo16) * DH_ + hi * 8;
    const unsigned short* kp1 = kp0 + 16 * DH_;
    kf[buf][0].q[0] = *(const uint4*)(kp0);      kf[buf][0].q[1] = *(const uint4*)(kp0 + 16);
    kf[buf][1].q[0] = *(const uint4*)(kp0 + 32); kf[buf][1].q[1] = *(const uint4*)(kp0 + 48);
    kf[buf][2].q[0] = *(const uint4*)(kp1);      kf[buf][2].q[1] = *(const uint4*)(kp1 + 16);
    kf[buf][3].q[0] = *(const uint4*)(kp1 + 32); kf[buf][3].q[1] = *(const uint4*)(kp1 + 48);
  };

  auto step = [&](int buf, int kb) {
    // issue next K block + this block's V early; softmax hides the latency
    if (kb + 32 < S_) load_k(buf ^ 1, kb + 32);
    FragAB bv[4];
#pragma unroll
    for (int f = 0; f < 4; ++f) {
      const unsigned short* vp = vbh + (f * 16 + lo16) * S_ + kb + hi * 8;
      bv[f].q[0] = *(const uint4*)(vp);
      bv[f].q[1] = *(const uint4*)(vp + 16);
    }
    if (kb + 64 < S_)  // gfx1250 global_prefetch of the V block after next
      __builtin_prefetch(vbh + lo16 * S_ + kb + 64, 0, 1);

    // ---- scores: two 16x16 C tiles
    v8f c0 = {}, c1 = {};
    c0 = wmma_bf16(aq0, kf[buf][0], c0);
    c0 = wmma_bf16(aq1, kf[buf][1], c0);
    c1 = wmma_bf16(aq0, kf[buf][2], c1);
    c1 = wmma_bf16(aq1, kf[buf][3], c1);

    // ---- online softmax update (row stats shared by 16-lane halves)
#pragma unroll
    for (int r = 0; r < 8; ++r) {
      const float v0 = c0[r], v1 = c1[r];
      const float mblk = rmax16(fmaxf(v0, v1));
      const float mnew = fmaxf(mrow[r], mblk);
      const float alpha = __expf(mrow[r] - mnew);
      const float p0 = __expf(v0 - mnew);
      const float p1 = __expf(v1 - mnew);
      lrow[r] = lrow[r] * alpha + rsum16(p0 + p1);
      mrow[r] = mnew;
#pragma unroll
      for (int f = 0; f < 4; ++f) acc[f][r] *= alpha;
      const int prow = r + (hi << 3);
      lds_p[wid][prow * 32 + lo16]      = (unsigned short)f2bf(p0);
      lds_p[wid][prow * 32 + 16 + lo16] = (unsigned short)f2bf(p1);
    }
    __syncthreads();

    // ---- P tile back as A-fragment (16 rows x 32 keys)
    FragAB ap;
    const unsigned short* pp = &lds_p[wid][lo16 * 32];
    ap.q[0] = *(const uint4*)(pp + krun);
    ap.q[1] = *(const uint4*)(pp + krun + 16);
    __syncthreads();

    // ---- ctx += P @ V
#pragma unroll
    for (int f = 0; f < 4; ++f) acc[f] = wmma_bf16(ap, bv[f], acc[f]);
  };

  load_k(0, 0);
  for (int kb = 0; kb < S_; kb += 64) {
    step(0, kb);
    step(1, kb + 32);
  }

  // ---- finalize: ctx[b, s, h*64+dh] in bf16
  const int b = bh >> 4, h = bh & 15;
#pragma unroll
  for (int r = 0; r < 8; ++r) {
    const float inv = 1.0f / lrow[r];
    const int s = q0 + r + (hi << 3);
#pragma unroll
    for (int f = 0; f < 4; ++f) {
      const int col = h * 64 + f * 16 + lo16;
      Ctx[(b * S_ + s) * D_ + col] = (unsigned short)f2bf(acc[f][r] * inv);
    }
  }
}

// ---------------------------------------------------------------- output GEMM
__global__ void __launch_bounds__(256, 1)
gemm_out(const unsigned short* __restrict__ X,
         const unsigned short* __restrict__ W,
         const float* __restrict__ bias,
         float* __restrict__ Y) {
  const int wave = (blockIdx.x * blockDim.x + threadIdx.x) >> 5;
  const int lane = threadIdx.x & 31;
  const int lo16 = lane & 15;
  const int hi   = lane >> 4;
  const int tn = wave & 15;
  const int tm = wave >> 4;

  v8f acc[2][4] = {};
  gemm_acc_32x64(X, W, tm, tn, lane, acc);

#pragma unroll
  for (int mi = 0; mi < 2; ++mi) {
#pragma unroll
    for (int j = 0; j < 4; ++j) {
      const int n = tn * 64 + j * 16 + lo16;
      const float bsc = bias[n];
#pragma unroll
      for (int r = 0; r < 8; ++r) {
        const int m = tm * 32 + mi * 16 + r + (hi << 3);
        Y[m * D_ + n] = acc[mi][j][r] + bsc;
      }
    }
  }
}

// ---------------------------------------------------------------- launch
extern "C" void kernel_launch(void* const* d_in, const int* in_sizes, int n_in,
                              void* d_out, int out_size, void* d_ws, size_t ws_size,
                              hipStream_t stream) {
  const float* query = (const float*)d_in[0];
  // d_in[1] = mask (mathematical no-op in reference)
  const float* q_w = (const float*)d_in[2];
  const float* q_b = (const float*)d_in[3];
  const float* k_w = (const float*)d_in[4];
  const float* k_b = (const float*)d_in[5];
  const float* v_w = (const float*)d_in[6];
  const float* v_b = (const float*)d_in[7];
  const float* out_w = (const float*)d_in[8];
  const float* out_b = (const float*)d_in[9];
  float* out = (float*)d_out;

  unsigned short* ws  = (unsigned short*)d_ws;
  unsigned short* Xbf = ws;                        // 4096x1024
  unsigned short* Wqb = Xbf + (size_t)M_TOT * D_;  // 1024x1024 each
  unsigned short* Wkb = Wqb + (size_t)D_ * D_;
  unsigned short* Wvb = Wkb + (size_t)D_ * D_;
  unsigned short* Wob = Wvb + (size_t)D_ * D_;
  unsigned short* Qh  = Wob + (size_t)D_ * D_;     // [B,H,S,DH]
  unsigned short* Kh  = Qh + (size_t)M_TOT * D_;   // [B,H,S,DH]
  unsigned short* Vt  = Kh + (size_t)M_TOT * D_;   // [B,H,DH,S]
  unsigned short* Ctx = Vt + (size_t)M_TOT * D_;   // [B,S,D]

  cvt_f32_bf16<<<(M_TOT * D_) / 2048, 256, 0, stream>>>(query, Xbf, M_TOT * D_);
  cvt_f32_bf16<<<(D_ * D_) / 2048, 256, 0, stream>>>(q_w, Wqb, D_ * D_);
  cvt_f32_bf16<<<(D_ * D_) / 2048, 256, 0, stream>>>(k_w, Wkb, D_ * D_);
  cvt_f32_bf16<<<(D_ * D_) / 2048, 256, 0, stream>>>(v_w, Wvb, D_ * D_);
  cvt_f32_bf16<<<(D_ * D_) / 2048, 256, 0, stream>>>(out_w, Wob, D_ * D_);

  // 2048 wave-tiles (128 m-tiles x 16 n-tiles), 8 waves/block, y = {Q,K,V}
  gemm_qkv<<<dim3(256, 3), 256, 0, stream>>>(Xbf, Wqb, Wkb, Wvb,
                                             q_b, k_b, v_b, Qh, Kh, Vt);
  // 32 head-batches x 128 q-tiles = 4096 waves
  flash_attn<<<512, 256, 0, stream>>>(Qh, Kh, Vt, Ctx);
  gemm_out<<<256, 256, 0, stream>>>(Ctx, Wob, out_b, out);
}
